// RWKV_Block_54485955117238
// MI455X (gfx1250) — compile-verified, matrix operand order checked by reading, not run on hardware
//
#include <hip/hip_runtime.h>

// ---------------------------------------------------------------------------
// RWKV block for MI455X (gfx1250): bf16 WMMA GEMMs (async-LDS double-buffered)
// + serial WKV scan.
// ---------------------------------------------------------------------------

typedef __bf16 bf16;
typedef bf16  v16bf __attribute__((ext_vector_type(16)));
typedef bf16  v8bf  __attribute__((ext_vector_type(8)));
typedef float v8f   __attribute__((ext_vector_type(8)));
typedef int   v4i   __attribute__((ext_vector_type(4)));

enum { EPI_F32 = 0, EPI_SIG = 1, EPI_RESID = 2, EPI_RELU2BF = 3, EPI_FINAL = 4 };

#if defined(__gfx1250__) && __has_builtin(__builtin_amdgcn_global_load_async_to_lds_b128)
#define USE_ASYNC_LDS 1
#else
#define USE_ASYNC_LDS 0
#endif

#if USE_ASYNC_LDS
typedef __attribute__((address_space(1))) v4i* gptr_v4i;
typedef __attribute__((address_space(3))) v4i* lptr_v4i;

__device__ __forceinline__ void async_g2l_b128(const bf16* g, bf16* l) {
    __builtin_amdgcn_global_load_async_to_lds_b128((gptr_v4i)g, (lptr_v4i)l, 0,
                                                   0);
}
template <int N>
__device__ __forceinline__ void wait_asynccnt() {
#if __has_builtin(__builtin_amdgcn_s_wait_asynccnt)
    __builtin_amdgcn_s_wait_asynccnt((short)N);
#else
    asm volatile("s_wait_asynccnt %0" ::"i"(N) : "memory");
#endif
}
#endif

// ---------------------------------------------------------------------------
// bf16 WMMA GEMM: C[M,N] = A[M,K] @ B[K,N], B supplied transposed as Bt[N,K]
// so B-fragment lane loads mirror the A-fragment layout.
// Block tile 128x128, BK=32 (one v_wmma_f32_16x16x32_bf16 K-step).
// 4 waves (2x2), wave tile 64x64 = 4x4 WMMA tiles -> 16 WMMA / 16 ds_load_b128.
// Double-buffered LDS fed by GLOBAL_LOAD_ASYNC_TO_LDS_B128 (ASYNCcnt), with
// the last iteration peeled so the hot loop has one unconditional
// s_wait_asynccnt 8 and no branch logic.
// ---------------------------------------------------------------------------
template <int EPI>
__global__ __launch_bounds__(128) void gemm_bf16_wmma(
    const bf16* __restrict__ A,    // [M,K] bf16 row-major
    const bf16* __restrict__ Bt,   // [N,K] bf16 row-major (B transposed)
    int M, int N, int K,
    float* __restrict__ Cf,        // f32 output
    bf16* __restrict__ Cbf,        // bf16 output (EPI_RELU2BF)
    const float* __restrict__ X,   // residual input (EPI_RESID/FINAL)
    const float* __restrict__ R)   // gate input (EPI_FINAL)
{
    constexpr int BM = 128, BN = 128, BK = 32, LDT = BK + 8;  // pad -> 40

    __shared__ __align__(16) bf16 As[2][BM * LDT];
    __shared__ __align__(16) bf16 Bs[2][BN * LDT];

    const int tid  = threadIdx.x;
    const int lane = tid & 31;
    const int wid  = tid >> 5;   // 0..3
    const int wm   = wid & 1;    // 64-row slab
    const int wn   = wid >> 1;   // 64-col slab
    const int gm0  = blockIdx.y * BM;
    const int gn0  = blockIdx.x * BN;

    const int l16   = lane & 15;
    const int lhalf = lane >> 4;  // 0/1
    const int klo   = lhalf * 8;  // K-chunk base within fragment

    v8f acc[4][4];
#pragma unroll
    for (int mi = 0; mi < 4; ++mi)
#pragma unroll
        for (int ni = 0; ni < 4; ++ni) acc[mi][ni] = (v8f)0.0f;

    // Per-thread tile-copy chunks: 512 16B-chunks per 128x32 tile, 4 per thread.
    int crow[4], ccol[4], loff[4];
#pragma unroll
    for (int i = 0; i < 4; ++i) {
        int c = tid + i * 128;
        crow[i] = c >> 2;
        ccol[i] = (c & 3) * 8;
        loff[i] = crow[i] * LDT + ccol[i];
    }

    const int nk = K / BK;

    // One WMMA K-step from LDS buffer p.
    auto compute_tile = [&](int p) {
        v16bf af[4], bfg[4];
        // Fragment loads per ISA 16-bit A layout: lanes 0-15 hold
        // K[0..7]+K[16..23]; lanes 16-31 hold K[8..15]+K[24..31].
#pragma unroll
        for (int mi = 0; mi < 4; ++mi) {
            int rr = (wm * 64 + mi * 16 + l16) * LDT + klo;
            v8bf lo = *(const v8bf*)&As[p][rr];
            v8bf hi = *(const v8bf*)&As[p][rr + 16];
            af[mi] = __builtin_shufflevector(lo, hi, 0, 1, 2, 3, 4, 5, 6, 7, 8,
                                             9, 10, 11, 12, 13, 14, 15);
        }
#pragma unroll
        for (int ni = 0; ni < 4; ++ni) {
            int rr = (wn * 64 + ni * 16 + l16) * LDT + klo;
            v8bf lo = *(const v8bf*)&Bs[p][rr];
            v8bf hi = *(const v8bf*)&Bs[p][rr + 16];
            bfg[ni] = __builtin_shufflevector(lo, hi, 0, 1, 2, 3, 4, 5, 6, 7, 8,
                                              9, 10, 11, 12, 13, 14, 15);
        }
#pragma unroll
        for (int mi = 0; mi < 4; ++mi)
#pragma unroll
            for (int ni = 0; ni < 4; ++ni)
                acc[mi][ni] = __builtin_amdgcn_wmma_f32_16x16x32_bf16(
                    false, af[mi], false, bfg[ni], (short)0, acc[mi][ni], false,
                    false);
    };

#if USE_ASYNC_LDS
    // Persistent per-thread global pointers, bumped by BK per tile.
    const bf16* ga[4];
    const bf16* gb[4];
#pragma unroll
    for (int i = 0; i < 4; ++i) {
        ga[i] = &A[(size_t)(gm0 + crow[i]) * K + ccol[i]];
        gb[i] = &Bt[(size_t)(gn0 + crow[i]) * K + ccol[i]];
    }

    // Prologue: async-load tile 0 into buffer 0.
#pragma unroll
    for (int i = 0; i < 4; ++i) {
        async_g2l_b128(ga[i], &As[0][loff[i]]);
        async_g2l_b128(gb[i], &Bs[0][loff[i]]);
        ga[i] += BK;
        gb[i] += BK;
    }

    // Steady state: issue tile kt+1, wait for tile kt, compute tile kt.
    for (int kt = 0; kt < nk - 1; ++kt) {
        const int pn = (kt + 1) & 1;
#pragma unroll
        for (int i = 0; i < 4; ++i) {
            async_g2l_b128(ga[i], &As[pn][loff[i]]);
            async_g2l_b128(gb[i], &Bs[pn][loff[i]]);
            ga[i] += BK;
            gb[i] += BK;
        }
        wait_asynccnt<8>();  // tile kt's 8 async ops retired (in-order)
        __syncthreads();     // tile kt visible to all waves
        compute_tile(kt & 1);
        __syncthreads();     // buffer free for overwrite next iteration
    }

    // Peeled last tile.
    wait_asynccnt<0>();
    __syncthreads();
    compute_tile((nk - 1) & 1);
#else
    // Fallback: register-staged copy, single LDS buffer.
    for (int kt = 0; kt < nk; ++kt) {
        const int k0 = kt * BK;
        uint4 ra[4], rb[4];
#pragma unroll
        for (int i = 0; i < 4; ++i) {
            ra[i] = *(const uint4*)&A[(size_t)(gm0 + crow[i]) * K + k0 + ccol[i]];
            rb[i] = *(const uint4*)&Bt[(size_t)(gn0 + crow[i]) * K + k0 + ccol[i]];
        }
        __syncthreads();  // prior compute done before overwriting the buffer
#pragma unroll
        for (int i = 0; i < 4; ++i) {
            *(uint4*)&As[0][loff[i]] = ra[i];
            *(uint4*)&Bs[0][loff[i]] = rb[i];
        }
        __syncthreads();
        compute_tile(0);
    }
#endif

    // Epilogue. C/D layout: VGPR i -> M=i (lanes<16) / M=8+i (lanes>=16),
    // N = lane & 15.
#pragma unroll
    for (int mi = 0; mi < 4; ++mi)
#pragma unroll
        for (int ni = 0; ni < 4; ++ni)
#pragma unroll
            for (int i = 0; i < 8; ++i) {
                int row = gm0 + wm * 64 + mi * 16 + i + lhalf * 8;
                int col = gn0 + wn * 64 + ni * 16 + l16;
                size_t idx = (size_t)row * N + col;
                float v = acc[mi][ni][i];
                if constexpr (EPI == EPI_F32) {
                    Cf[idx] = v;
                } else if constexpr (EPI == EPI_SIG) {
                    Cf[idx] = 1.0f / (1.0f + __expf(-v));
                } else if constexpr (EPI == EPI_RESID) {
                    Cf[idx] = X[idx] + v;
                } else if constexpr (EPI == EPI_RELU2BF) {
                    float t = v > 0.0f ? v * v : 0.0f;
                    Cbf[idx] = (bf16)t;
                } else {  // EPI_FINAL: out = x1 + r * (kk @ cm_Wv)
                    Cf[idx] = X[idx] + R[idx] * v;
                }
            }
}

// ---------------------------------------------------------------------------
// LayerNorm over last dim (C), output bf16. One block per row.
// ---------------------------------------------------------------------------
__device__ __forceinline__ float wave_red_sum(float v) {
#pragma unroll
    for (int o = 16; o > 0; o >>= 1) v += __shfl_down(v, o, 32);
    return v;
}

__global__ __launch_bounds__(256) void layernorm_bf16(
    const float* __restrict__ x, const float* __restrict__ g,
    const float* __restrict__ b, bf16* __restrict__ y, int Cn)
{
    const int row = blockIdx.x;
    const int tid = threadIdx.x;
    const float* xr = x + (size_t)row * Cn;

    float s = 0.0f, s2 = 0.0f;
    for (int i = tid; i < Cn; i += 256) {
        float v = xr[i];
        s += v;
        s2 += v * v;
    }
    s = wave_red_sum(s);
    s2 = wave_red_sum(s2);

    __shared__ float part[16];
    __shared__ float stats[2];
    const int lane = tid & 31, wid = tid >> 5;
    if (lane == 0) {
        part[wid] = s;
        part[8 + wid] = s2;
    }
    __syncthreads();
    if (tid == 0) {
        float a = 0.0f, q = 0.0f;
#pragma unroll
        for (int i = 0; i < 8; ++i) {
            a += part[i];
            q += part[8 + i];
        }
        float mu = a / (float)Cn;
        float var = q / (float)Cn - mu * mu;
        stats[0] = mu;
        stats[1] = rsqrtf(var + 1e-5f);
    }
    __syncthreads();
    const float mu = stats[0], rstd = stats[1];
    bf16* yr = y + (size_t)row * Cn;
    for (int i = tid; i < Cn; i += 256)
        yr[i] = (bf16)((xr[i] - mu) * rstd * g[i] + b[i]);
}

// ---------------------------------------------------------------------------
// f32 [K,N] -> bf16 [N,K] transpose-convert (LDS tiled, both sides coalesced).
// ---------------------------------------------------------------------------
__global__ __launch_bounds__(256) void transpose_f32_to_bf16(
    const float* __restrict__ W, bf16* __restrict__ Wt, int Kd, int Nd)
{
    __shared__ float tile[32][33];
    const int n0 = blockIdx.x * 32, k0 = blockIdx.y * 32;
    const int tx = threadIdx.x & 31, ty = threadIdx.x >> 5;  // ty 0..7
#pragma unroll
    for (int i = 0; i < 32; i += 8)
        tile[ty + i][tx] = W[(size_t)(k0 + ty + i) * Nd + n0 + tx];
    __syncthreads();
#pragma unroll
    for (int i = 0; i < 32; i += 8)
        Wt[(size_t)(n0 + ty + i) * Kd + k0 + tx] = (bf16)tile[tx][ty + i];
}

// ---------------------------------------------------------------------------
// WKV scan: per (b, c) channel, wkv_t = w*wkv_{t-1} + k_t ; out = r_t * wkv_t.
// Coalesced across c; serial over T. Unrolled x8 to keep loads in flight.
// ---------------------------------------------------------------------------
__global__ __launch_bounds__(256) void wkv_scan(
    const float* __restrict__ k, const float* __restrict__ r,
    const float* __restrict__ td, bf16* __restrict__ rwkv, int Tn, int Cn)
{
    const int c = blockIdx.x * 256 + threadIdx.x;
    const int b = blockIdx.y;
    const float w = -__expf(td[c]);
    float st = 0.0f;
    size_t base = (size_t)b * Tn * Cn + c;
    for (int t = 0; t < Tn; t += 8) {
        float kv[8], rv[8];
#pragma unroll
        for (int j = 0; j < 8; ++j) {
            size_t idx = base + (size_t)(t + j) * Cn;
            kv[j] = k[idx];
            rv[j] = r[idx];
        }
#pragma unroll
        for (int j = 0; j < 8; ++j) {
            st = st * w + kv[j];
            rwkv[base + (size_t)(t + j) * Cn] = (bf16)(rv[j] * st);
        }
    }
}

// ---------------------------------------------------------------------------
// Host launcher
// ---------------------------------------------------------------------------
extern "C" void kernel_launch(void* const* d_in, const int* in_sizes, int n_in,
                              void* d_out, int out_size, void* d_ws,
                              size_t ws_size, hipStream_t stream) {
    (void)in_sizes; (void)n_in; (void)out_size; (void)ws_size;

    const float* x    = (const float*)d_in[0];
    const float* ln1g = (const float*)d_in[1];
    const float* ln1b = (const float*)d_in[2];
    const float* ln2g = (const float*)d_in[3];
    const float* ln2b = (const float*)d_in[4];
    const float* td   = (const float*)d_in[5];
    /* d_in[6] = time_first: only multiplies the zero initial state (unused) */
    const float* Wk   = (const float*)d_in[7];
    const float* Wv   = (const float*)d_in[8];
    const float* Wr   = (const float*)d_in[9];
    const float* cmWk = (const float*)d_in[10];
    const float* cmWr = (const float*)d_in[11];
    const float* cmWv = (const float*)d_in[12];
    float* out = (float*)d_out;

    constexpr int Bn = 4, Tn = 2048, Cn = 2048, Hn = 4 * Cn;
    constexpr int Mn = Bn * Tn;  // 8192 rows

    char* wsb = (char*)d_ws;
    size_t off = 0;
    auto take = [&](size_t bytes) -> char* {
        char* p = wsb + off;
        off = (off + bytes + 255) & ~(size_t)255;
        return p;
    };
    bf16* WkT    = (bf16*)take((size_t)Cn * Cn * sizeof(bf16));
    bf16* WvT    = (bf16*)take((size_t)Cn * Cn * sizeof(bf16));
    bf16* WrT    = (bf16*)take((size_t)Cn * Cn * sizeof(bf16));
    bf16* cmWkT  = (bf16*)take((size_t)Hn * Cn * sizeof(bf16));  // [H,C]
    bf16* cmWvT  = (bf16*)take((size_t)Cn * Hn * sizeof(bf16));  // [C,H]
    bf16* cmWrT  = (bf16*)take((size_t)Cn * Cn * sizeof(bf16));
    bf16* xnBF   = (bf16*)take((size_t)Mn * Cn * sizeof(bf16));
    float* kF    = (float*)take((size_t)Mn * Cn * sizeof(float));
    float* rF    = (float*)take((size_t)Mn * Cn * sizeof(float));
    bf16* rwkvBF = (bf16*)take((size_t)Mn * Cn * sizeof(bf16));
    float* x1F   = (float*)take((size_t)Mn * Cn * sizeof(float));
    bf16* kkBF   = (bf16*)take((size_t)Mn * Hn * sizeof(bf16));

    const dim3 blk256(256);
    const dim3 blk128(128);

    // 1) Weight transpose-converts: W[K,N] f32 -> Wt[N,K] bf16.
    transpose_f32_to_bf16<<<dim3(Cn / 32, Cn / 32), blk256, 0, stream>>>(Wk, WkT, Cn, Cn);
    transpose_f32_to_bf16<<<dim3(Cn / 32, Cn / 32), blk256, 0, stream>>>(Wv, WvT, Cn, Cn);
    transpose_f32_to_bf16<<<dim3(Cn / 32, Cn / 32), blk256, 0, stream>>>(Wr, WrT, Cn, Cn);
    transpose_f32_to_bf16<<<dim3(Hn / 32, Cn / 32), blk256, 0, stream>>>(cmWk, cmWkT, Cn, Hn);
    transpose_f32_to_bf16<<<dim3(Cn / 32, Hn / 32), blk256, 0, stream>>>(cmWv, cmWvT, Hn, Cn);
    transpose_f32_to_bf16<<<dim3(Cn / 32, Cn / 32), blk256, 0, stream>>>(cmWr, cmWrT, Cn, Cn);

    // 2) LN1 -> bf16 activations
    layernorm_bf16<<<dim3(Mn), blk256, 0, stream>>>(x, ln1g, ln1b, xnBF, Cn);

    // 3) k = xn @ Wk ; r = sigmoid(xn @ Wr)
    gemm_bf16_wmma<EPI_F32><<<dim3(Cn / 128, Mn / 128), blk128, 0, stream>>>(
        xnBF, WkT, Mn, Cn, Cn, kF, nullptr, nullptr, nullptr);
    gemm_bf16_wmma<EPI_SIG><<<dim3(Cn / 128, Mn / 128), blk128, 0, stream>>>(
        xnBF, WrT, Mn, Cn, Cn, rF, nullptr, nullptr, nullptr);

    // 4) WKV scan -> rwkv (bf16)
    wkv_scan<<<dim3(Cn / 256, Bn), blk256, 0, stream>>>(kF, rF, td, rwkvBF, Tn, Cn);

    // 5) x1 = x + rwkv @ Wv
    gemm_bf16_wmma<EPI_RESID><<<dim3(Cn / 128, Mn / 128), blk128, 0, stream>>>(
        rwkvBF, WvT, Mn, Cn, Cn, x1F, nullptr, x, nullptr);

    // 6) LN2 -> bf16
    layernorm_bf16<<<dim3(Mn), blk256, 0, stream>>>(x1F, ln2g, ln2b, xnBF, Cn);

    // 7) kk = relu(xn2 @ cm_Wk)^2  (bf16, [M,H])
    gemm_bf16_wmma<EPI_RELU2BF><<<dim3(Hn / 128, Mn / 128), blk128, 0, stream>>>(
        xnBF, cmWkT, Mn, Hn, Cn, nullptr, kkBF, nullptr, nullptr);

    // 8) r2 = sigmoid(xn2 @ cm_Wr)  (reuses rF)
    gemm_bf16_wmma<EPI_SIG><<<dim3(Cn / 128, Mn / 128), blk128, 0, stream>>>(
        xnBF, cmWrT, Mn, Cn, Cn, rF, nullptr, nullptr, nullptr);

    // 9) out = x1 + r2 * (kk @ cm_Wv)   (K = H = 8192)
    gemm_bf16_wmma<EPI_FINAL><<<dim3(Cn / 128, Mn / 128), blk128, 0, stream>>>(
        kkBF, cmWvT, Mn, Cn, Hn, out, nullptr, x1F, rF);
}
